// HearInt_46703474377037
// MI455X (gfx1250) — matrix operand order repaired and checked by reading, not compiled, not run on hardware
//
#include <hip/hip_runtime.h>

// GAT-style typed-edge attention for MI455X (gfx1250, wave32, WMMA).
// B=512, N=64, D=128. One workgroup (256 thr = 8 waves) per batch element.
// Register-resident A-fragments: each B-fragment load feeds 2 WMMAs.

typedef __attribute__((ext_vector_type(8)))  _Float16 v8h;
typedef __attribute__((ext_vector_type(16))) _Float16 v16h;
typedef __attribute__((ext_vector_type(8)))  float    v8f;

#define BATCH 512
#define NN    64
#define DD    128
#define HF_S  136   // f16 stride of Hf rows  (272B = 17*16B, conflict-free b128)
#define HT_S  72    // f16 stride of HfT rows (144B = 9*16B)
#define ES_S  72    // f16 stride of E rows
#define AL_S  72    // f16 stride of alpha rows
#define NEGF  (-9.0e15f)

__global__ __launch_bounds__(256) void gat_typed_attn_kernel(
    const float* __restrict__ hidden,   // [512][64][128] f32
    const int*   __restrict__ adj,      // [512][64][64]  i32
    const float* __restrict__ Amat,     // [4][128]       f32
    float*       __restrict__ out)      // [512][64][128] f32
{
    __shared__ __align__(16) _Float16 Hf [NN * HF_S];      // 17408 B
    __shared__ __align__(16) _Float16 HfT[DD * HT_S];      // 18432 B
    __shared__ __align__(16) _Float16 Af [4  * DD];        //  1024 B
    __shared__ __align__(16) _Float16 Es [4 * NN * ES_S];  // 36864 B
    __shared__ __align__(16) _Float16 Al [NN * AL_S];      //  9216 B

    const int tid  = threadIdx.x;
    const int wave = tid >> 5;
    const int lane = tid & 31;
    const int half = lane >> 4;     // 0: lanes 0-15, 1: lanes 16-31
    const int l    = lane & 15;
    const int b    = blockIdx.x;

    const float* hB = hidden + (size_t)b * (NN * DD);

    // ---------------- Phase 1: H[b] f32 -> f16 LDS (row + transposed) -------
    {
        const int i = tid >> 2;        // row 0..63
        const int q = tid & 3;         // column quarter
        const float4* src = (const float4*)(hB + i * DD + q * 32);
#pragma unroll
        for (int t = 0; t < 8; ++t) {
            float4 v = src[t];
            int c = q * 32 + t * 4;
            _Float16 h0 = (_Float16)v.x, h1 = (_Float16)v.y;
            _Float16 h2 = (_Float16)v.z, h3 = (_Float16)v.w;
            Hf[i * HF_S + c + 0] = h0;  Hf[i * HF_S + c + 1] = h1;
            Hf[i * HF_S + c + 2] = h2;  Hf[i * HF_S + c + 3] = h3;
            HfT[(c + 0) * HT_S + i] = h0;  HfT[(c + 1) * HT_S + i] = h1;
            HfT[(c + 2) * HT_S + i] = h2;  HfT[(c + 3) * HT_S + i] = h3;
        }
        if (tid < DD) {
#pragma unroll
            for (int k = 0; k < 4; ++k)
                Af[k * DD + tid] = (_Float16)Amat[k * DD + tid];
        }
        // warm the cache path for the adj tile this thread gathers in phase 3
        __builtin_prefetch(adj + (size_t)b * (NN * NN) + (tid >> 2) * NN + (tid & 3) * 16, 0, 0);
    }
    __syncthreads();

    // ---------------- Phase 2: E_k = leakyrelu((H .* A_k) @ H^T) ------------
    // Wave w owns edge-type k = w>>1 and row-tiles ti = {2*(w&1), 2*(w&1)+1}.
    // A-fragments (pre-scaled by A_k) are held in registers across the tj sweep;
    // each B-fragment load feeds two WMMAs.
    {
        const int k      = wave >> 1;
        const int tibase = (wave & 1) * 2;

        v16h afr[2][4];                         // [ti'][kk] : 64 VGPRs
#pragma unroll
        for (int s = 0; s < 2; ++s) {
            const int row = (tibase + s) * 16 + l;
#pragma unroll
            for (int kk = 0; kk < 4; ++kk) {
                const int ac = kk * 32;
                const v8h alo = *(const v8h*)&Hf[row * HF_S + ac + half * 8];
                const v8h ahi = *(const v8h*)&Hf[row * HF_S + ac + 16 + half * 8];
                const v8h flo = *(const v8h*)&Af[k * DD + ac + half * 8];
                const v8h fhi = *(const v8h*)&Af[k * DD + ac + 16 + half * 8];
#pragma unroll
                for (int t = 0; t < 8; ++t) {
                    afr[s][kk][t]     = alo[t] * flo[t];
                    afr[s][kk][t + 8] = ahi[t] * fhi[t];
                }
            }
        }

#pragma unroll 1
        for (int tj = 0; tj < 4; ++tj) {
            v8f c0 = {}, c1 = {};
#pragma unroll
            for (int kk = 0; kk < 4; ++kk) {
                const int ac = kk * 32;
                // B-fragment: column tj*16+l, K = half*8 .. +7 and 16+half*8 .. +7
                const v8h b0 = *(const v8h*)&Hf[(tj * 16 + l) * HF_S + ac + half * 8];
                const v8h b1 = *(const v8h*)&Hf[(tj * 16 + l) * HF_S + ac + 16 + half * 8];
                v16h bb;
#pragma unroll
                for (int t = 0; t < 8; ++t) { bb[t] = b0[t]; bb[t + 8] = b1[t]; }
                c0 = __builtin_amdgcn_wmma_f32_16x16x32_f16(false, afr[0][kk], false, bb,
                                                            (short)0, c0, false, false);
                c1 = __builtin_amdgcn_wmma_f32_16x16x32_f16(false, afr[1][kk], false, bb,
                                                            (short)0, c1, false, false);
            }
            // leaky-relu + store tiles: VGPR r -> (M = r + 8*half, N = l)
#pragma unroll
            for (int r = 0; r < 8; ++r) {
                float v0 = c0[r];
                v0 = v0 > 0.0f ? v0 : 0.2f * v0;
                Es[(k * NN + tibase * 16 + r + half * 8) * ES_S + tj * 16 + l] = (_Float16)v0;
                float v1 = c1[r];
                v1 = v1 > 0.0f ? v1 : 0.2f * v1;
                Es[(k * NN + (tibase + 1) * 16 + r + half * 8) * ES_S + tj * 16 + l] = (_Float16)v1;
            }
        }
    }
    __syncthreads();

    // ---------------- Phase 3: typed-edge select + row softmax --------------
    {
        const int i = tid >> 2;   // row
        const int q = tid & 3;    // 16-wide j segment
        const int* arow = adj + (size_t)b * (NN * NN) + i * NN + q * 16;
        float vals[16];
        float m = -3.4e38f;
#pragma unroll
        for (int t = 0; t < 16; ++t) {
            const unsigned at = (unsigned)(arow[t] - 1);
            float v = NEGF;
            if (at < 4u) v = (float)Es[(at * NN + i) * ES_S + q * 16 + t];
            vals[t] = v;
            m = fmaxf(m, v);
        }
        m = fmaxf(m, __shfl_xor(m, 1, 4));
        m = fmaxf(m, __shfl_xor(m, 2, 4));
        float s = 0.0f;
#pragma unroll
        for (int t = 0; t < 16; ++t) {
            const float e = __expf(vals[t] - m);
            vals[t] = e;
            s += e;
        }
        s += __shfl_xor(s, 1, 4);
        s += __shfl_xor(s, 2, 4);
        const float rinv = __frcp_rn(s);
#pragma unroll
        for (int t = 0; t < 16; ++t)
            Al[i * AL_S + q * 16 + t] = (_Float16)(vals[t] * rinv);
    }
    __syncthreads();

    // ---------------- Phase 4: out = alpha @ H ------------------------------
    // Wave w owns ti = w>>1, td = (w&1)*4 .. +3. Alpha A-fragments held in
    // registers across the td sweep.
    {
        const int ti  = wave >> 1;
        const int tdb = (wave & 1) * 4;

        v16h aal[2];                            // [jc] : 16 VGPRs
#pragma unroll
        for (int jc = 0; jc < 2; ++jc) {
            const int jb = jc * 32;
            const v8h alo = *(const v8h*)&Al[(ti * 16 + l) * AL_S + jb + half * 8];
            const v8h ahi = *(const v8h*)&Al[(ti * 16 + l) * AL_S + jb + 16 + half * 8];
#pragma unroll
            for (int t = 0; t < 8; ++t) { aal[jc][t] = alo[t]; aal[jc][t + 8] = ahi[t]; }
        }

#pragma unroll 1
        for (int t4 = 0; t4 < 4; ++t4) {
            const int td = tdb + t4;
            v8f c = {};
#pragma unroll
            for (int jc = 0; jc < 2; ++jc) {
                const int jb = jc * 32;
                // B-fragment from transposed copy: column d = td*16+l, K = j (half*8 .. +7, 16+half*8 .. +7)
                const v8h b0 = *(const v8h*)&HfT[(td * 16 + l) * HT_S + jb + half * 8];
                const v8h b1 = *(const v8h*)&HfT[(td * 16 + l) * HT_S + jb + 16 + half * 8];
                v16h bb;
#pragma unroll
                for (int t = 0; t < 8; ++t) { bb[t] = b0[t]; bb[t + 8] = b1[t]; }
                c = __builtin_amdgcn_wmma_f32_16x16x32_f16(false, aal[jc], false, bb,
                                                           (short)0, c, false, false);
            }
            float* orow = out + (size_t)b * (NN * DD) + td * 16 + l;
#pragma unroll
            for (int r = 0; r < 8; ++r)
                orow[(ti * 16 + r + half * 8) * DD] = c[r];
        }
    }
}

extern "C" void kernel_launch(void* const* d_in, const int* in_sizes, int n_in,
                              void* d_out, int out_size, void* d_ws, size_t ws_size,
                              hipStream_t stream) {
    const float* hidden = (const float*)d_in[0];
    const int*   adjm   = (const int*)d_in[1];
    const float* Amat   = (const float*)d_in[2];
    float*       outp   = (float*)d_out;
    (void)in_sizes; (void)n_in; (void)out_size; (void)d_ws; (void)ws_size;
    gat_typed_attn_kernel<<<dim3(BATCH), dim3(256), 0, stream>>>(hidden, adjm, Amat, outp);
}
